// Attention_9887014715893
// MI455X (gfx1250) — matrix operand order repaired and checked by reading, not compiled, not run on hardware
//
#include <hip/hip_runtime.h>
#include <hip/hip_bf16.h>
#include <math.h>

// ---------------------------------------------------------------------------
// Attention: x[1,4096,1024] @ w_qkv[1024,3072] -> flash attention -> out
// Stage 1: QKV GEMM with v_wmma_f32_16x16x32_bf16 (f32 -> bf16, f32 acc)
// Stage 2: flash attention; K/V tiles staged to LDS by the Tensor Data Mover
//          (tensor_load_to_lds + s_wait_tensorcnt), double buffered.
// ---------------------------------------------------------------------------

typedef __attribute__((ext_vector_type(16))) __bf16          v16bf;
typedef __attribute__((ext_vector_type(8)))  float           v8f;
typedef __attribute__((ext_vector_type(8)))  unsigned short  u16x8;
typedef __attribute__((ext_vector_type(16))) unsigned short  u16x16;
typedef __attribute__((ext_vector_type(4)))  unsigned short  u16x4;
typedef __attribute__((ext_vector_type(4)))  unsigned int    v4u;
typedef __attribute__((ext_vector_type(4)))  int             v4i;
typedef __attribute__((ext_vector_type(8)))  int             v8i;

#define S_LEN   4096
#define D_MODEL 1024
#define NH      16
#define HD      64
#define N3      3072
#define KVT     64          // kv tile width in flash kernel

static __device__ __forceinline__ unsigned short f2bf(float f) {
  unsigned u = __builtin_bit_cast(unsigned, f);
  u += 0x7FFFu + ((u >> 16) & 1u);           // round-to-nearest-even
  return (unsigned short)(u >> 16);
}

static __device__ __forceinline__ v16bf mkfrag(u16x8 lo, u16x8 hi) {
  u16x16 w = __builtin_shufflevector(lo, hi, 0,1,2,3,4,5,6,7,8,9,10,11,12,13,14,15);
  return __builtin_bit_cast(v16bf, w);
}

// --- Tensor Data Mover: 2-D tile load (bf16 elements) via D# descriptor ----
// g0: count=1 | lds_addr | global_addr | type=2
// g1: data_size=2B, tensor_dim0/1, tile_dim0/1, tensor_dim0_stride
static __device__ __forceinline__ void tdm_load_tile(
    unsigned lds_byte, unsigned long long gaddr,
    unsigned td0, unsigned td1, unsigned tile0, unsigned tile1,
    unsigned long long stride0)
{
  v4u g0;
  g0[0] = 1u;                                           // count = 1 (valid)
  g0[1] = lds_byte;                                     // LDS byte address
  g0[2] = (unsigned)gaddr;                              // global addr [31:0]
  g0[3] = (unsigned)((gaddr >> 32) & 0x1FFFFFFu) | (2u << 30);  // [56:32] | type=2
  v8i g1;
  g1[0] = (int)(1u << 16);                              // data_size = 1 -> 2 bytes
  g1[1] = (int)((td0 & 0xFFFFu) << 16);                 // tensor_dim0 [15:0]
  g1[2] = (int)((td0 >> 16) | ((td1 & 0xFFFFu) << 16)); // td0[31:16] | td1[15:0]
  g1[3] = (int)((td1 >> 16) | (tile0 << 16));           // td1[31:16] | tile_dim0
  g1[4] = (int)(tile1 & 0xFFFFu);                       // tile_dim1 (tile_dim2 = 0)
  g1[5] = (int)(unsigned)stride0;                       // tensor_dim0_stride [31:0]
  g1[6] = (int)((unsigned)(stride0 >> 32) & 0xFFFFu);   // stride0[47:32]
  g1[7] = 0;
  v4i z = {};
#if __has_include(<hip/amd_detail/amd_gfx1250_TDM.h>)
  v8i z8 = {};
  __builtin_amdgcn_tensor_load_to_lds(g0, g1, z, z, z8, 0);   // therock: 6-arg
#else
  __builtin_amdgcn_tensor_load_to_lds(g0, g1, z, z, 0);       // ROCm 7.2: 5-arg
#endif
}

// ---------------------------------------------------------------------------
// Kernel 1: qkv = x @ w_qkv  ->  Qb [H][S][E] (pre-scaled 1/8), Kb [H][S][E],
// Vt [H][E][S], all bf16. 256 thr, tile 128x64, BK=32, 2x2 WMMA per wave.
// ---------------------------------------------------------------------------
__global__ __launch_bounds__(256) void qkv_gemm_kernel(
    const float* __restrict__ X, const float* __restrict__ W,
    unsigned short* __restrict__ Qb, unsigned short* __restrict__ Kb,
    unsigned short* __restrict__ Vt)
{
  __shared__ unsigned short Xs[128 * 32];   // [row][k]  bf16
  __shared__ unsigned short WsT[64 * 32];   // [col][k]  bf16 (transposed)

  const int t    = threadIdx.x;
  const int lane = t & 31;
  const int wave = t >> 5;
  const int wm   = wave >> 1;
  const int wn   = wave & 1;
  const int tileM = blockIdx.y * 128;
  const int tileN = blockIdx.x * 64;
  const int hl   = (lane < 16) ? 0 : 1;
  const int l15  = lane & 15;

  v8f zero = {};
  v8f acc[2][2];
  acc[0][0] = zero; acc[0][1] = zero; acc[1][0] = zero; acc[1][1] = zero;

  for (int k0 = 0; k0 < D_MODEL; k0 += 32) {
    #pragma unroll
    for (int i = 0; i < 4; i++) {
      int f   = t + i * 256;
      int row = f >> 3, c4 = f & 7;
      const float4 v = *(const float4*)&X[(size_t)(tileM + row) * D_MODEL + k0 + c4 * 4];
      u16x4 p;
      p[0] = f2bf(v.x); p[1] = f2bf(v.y); p[2] = f2bf(v.z); p[3] = f2bf(v.w);
      *(u16x4*)&Xs[row * 32 + c4 * 4] = p;
    }
    #pragma unroll
    for (int i = 0; i < 2; i++) {
      int f  = t + i * 256;
      int kr = f >> 4, c4 = f & 15;
      const float4 v = *(const float4*)&W[(size_t)(k0 + kr) * N3 + tileN + c4 * 4];
      WsT[(c4 * 4 + 0) * 32 + kr] = f2bf(v.x);
      WsT[(c4 * 4 + 1) * 32 + kr] = f2bf(v.y);
      WsT[(c4 * 4 + 2) * 32 + kr] = f2bf(v.z);
      WsT[(c4 * 4 + 3) * 32 + kr] = f2bf(v.w);
    }
    __syncthreads();

    v16bf af[2], bfr[2];
    #pragma unroll
    for (int mi = 0; mi < 2; mi++) {
      int base = (wm * 32 + mi * 16 + l15) * 32 + hl * 8;
      af[mi] = mkfrag(*(const u16x8*)&Xs[base], *(const u16x8*)&Xs[base + 16]);
    }
    #pragma unroll
    for (int ni = 0; ni < 2; ni++) {
      int base = (wn * 32 + ni * 16 + l15) * 32 + hl * 8;
      bfr[ni] = mkfrag(*(const u16x8*)&WsT[base], *(const u16x8*)&WsT[base + 16]);
    }
    #pragma unroll
    for (int mi = 0; mi < 2; mi++)
      #pragma unroll
      for (int ni = 0; ni < 2; ni++)
        acc[mi][ni] = __builtin_amdgcn_wmma_f32_16x16x32_bf16(
            false, af[mi], false, bfr[ni], (short)0, acc[mi][ni], false, false);
    __syncthreads();
  }

  #pragma unroll
  for (int mi = 0; mi < 2; mi++) {
    int sbase = tileM + wm * 32 + mi * 16 + hl * 8;
    #pragma unroll
    for (int ni = 0; ni < 2; ni++) {
      int colbase = tileN + wn * 32 + ni * 16;
      int k3 = colbase >> 10;
      int h  = (colbase & 1023) >> 6;
      int e  = (colbase & 63) + l15;
      if (k3 == 2) {
        u16x8 p;
        #pragma unroll
        for (int r = 0; r < 8; r++) p[r] = f2bf(acc[mi][ni][r]);
        *(u16x8*)&Vt[((size_t)(h * HD + e)) * S_LEN + sbase] = p;
      } else {
        unsigned short* dst = (k3 == 0) ? Qb : Kb;
        float scl = (k3 == 0) ? 0.125f : 1.0f;
        #pragma unroll
        for (int r = 0; r < 8; r++)
          dst[((size_t)h * S_LEN + sbase + r) * HD + e] = f2bf(acc[mi][ni][r] * scl);
      }
    }
  }
}

// ---------------------------------------------------------------------------
// Kernel 2: flash attention. 128 thr = 4 waves, each wave owns 16 Q rows.
// KV streamed in 64-wide tiles staged into LDS by the TDM (double buffered);
// 16 WMMAs per tile (8 score + 8 PV).
// ---------------------------------------------------------------------------
__global__ __launch_bounds__(128) void flash_attn_kernel(
    const unsigned short* __restrict__ Qb, const unsigned short* __restrict__ Kb,
    const unsigned short* __restrict__ Vt, const float* __restrict__ mask,
    float* __restrict__ out)
{
  __shared__ unsigned short Kt[2][KVT * HD];    // [buf][kv][e]  8KB each
  __shared__ unsigned short Vs[2][HD * KVT];    // [buf][e][kv]  8KB each
  __shared__ unsigned short Ps[4][16 * KVT];    // per-wave P bounce, 2KB each

  const int lane = threadIdx.x & 31;
  const int wave = threadIdx.x >> 5;
  const int h    = blockIdx.y;
  const int q0   = blockIdx.x * 64 + wave * 16;
  const int hl   = (lane < 16) ? 0 : 1;
  const int l15  = lane & 15;
  const int rowbase = q0 + hl * 8;              // C-layout: VGPR r -> row rowbase+r

  const unsigned short* kbh = &Kb[(size_t)h * S_LEN * HD];
  const unsigned short* vth = &Vt[(size_t)h * HD * S_LEN];

  // --- wave 0 kicks off the TDM for tile 0 immediately ---
  if (wave == 0) {
    // K tile: 64 rows x 64 e, rows contiguous => one 1-D 8KB move
    tdm_load_tile((unsigned)(size_t)&Kt[0][0], (unsigned long long)(size_t)kbh,
                  KVT * HD, 1, KVT * HD, 0, KVT * HD);
    // V tile: 64 e-rows x 64 kv, row stride = S (elements)
    tdm_load_tile((unsigned)(size_t)&Vs[0][0], (unsigned long long)(size_t)vth,
                  S_LEN, HD, KVT, HD, S_LEN);
  }

  // --- Q A-fragments, resident for the whole sweep (K dim 64 = 2 steps) ---
  v16bf aq[2];
  {
    const unsigned short* qp = &Qb[((size_t)h * S_LEN + q0 + l15) * HD];
    #pragma unroll
    for (int kk = 0; kk < 2; kk++) {
      int b = kk * 32 + hl * 8;
      aq[kk] = mkfrag(*(const u16x8*)&qp[b], *(const u16x8*)&qp[b + 16]);
    }
  }

  float m[8], l[8];
  v8f zero = {};
  v8f o[4];
  #pragma unroll
  for (int r = 0; r < 8; r++) { m[r] = -INFINITY; l[r] = 0.0f; }
  #pragma unroll
  for (int tt = 0; tt < 4; tt++) o[tt] = zero;

  const int NIT = S_LEN / KVT;                  // 64 iterations
  for (int it = 0; it < NIT; ++it) {
    const int buf = it & 1;
    const int kv0 = it * KVT;

    if (wave == 0) __builtin_amdgcn_s_wait_tensorcnt(0);   // tile `it` landed
    __syncthreads();                            // LDS visible; prev compute done
    if (wave == 0 && it + 1 < NIT) {            // prefetch tile it+1 into other buf
      const int kvn = kv0 + KVT;
      tdm_load_tile((unsigned)(size_t)&Kt[buf ^ 1][0],
                    (unsigned long long)(size_t)&kbh[(size_t)kvn * HD],
                    KVT * HD, 1, KVT * HD, 0, KVT * HD);
      tdm_load_tile((unsigned)(size_t)&Vs[buf ^ 1][0],
                    (unsigned long long)(size_t)&vth[kvn],
                    S_LEN, HD, KVT, HD, S_LEN);
    }

    // ---- scores: 4 n-tiles, acc initialized with mask (score = Q'K + mask) ----
    v8f sc[4];
    #pragma unroll
    for (int n = 0; n < 4; n++) {
      int col = kv0 + n * 16 + l15;
      const float* mrow = &mask[(size_t)rowbase * S_LEN + col];
      v8f c;
      #pragma unroll
      for (int r = 0; r < 8; r++) c[r] = mrow[(size_t)r * S_LEN];
      #pragma unroll
      for (int kk = 0; kk < 2; kk++) {
        const unsigned short* kp = &Kt[buf][(n * 16 + l15) * HD + kk * 32 + hl * 8];
        v16bf bk = mkfrag(*(const u16x8*)kp, *(const u16x8*)(kp + 16));
        c = __builtin_amdgcn_wmma_f32_16x16x32_bf16(false, aq[kk], false, bk,
                                                    (short)0, c, false, false);
      }
      sc[n] = c;
    }

    // ---- online softmax ----
    float alpha[8];
    #pragma unroll
    for (int r = 0; r < 8; r++) {
      float rm = fmaxf(fmaxf(sc[0][r], sc[1][r]), fmaxf(sc[2][r], sc[3][r]));
      #pragma unroll
      for (int d = 8; d >= 1; d >>= 1) rm = fmaxf(rm, __shfl_xor(rm, d, 32));
      float mn = fmaxf(m[r], rm);
      alpha[r] = __expf(m[r] - mn);
      float rs = 0.0f;
      #pragma unroll
      for (int n = 0; n < 4; n++) {
        float p = __expf(sc[n][r] - mn);
        sc[n][r] = p;
        rs += p;
      }
      #pragma unroll
      for (int d = 8; d >= 1; d >>= 1) rs += __shfl_xor(rs, d, 32);
      l[r] = l[r] * alpha[r] + rs;
      m[r] = mn;
    }
    #pragma unroll
    for (int tt = 0; tt < 4; tt++)
      #pragma unroll
      for (int r = 0; r < 8; r++) o[tt][r] *= alpha[r];

    // ---- relayout P: C-layout -> A-layout via per-wave LDS bounce ----
    unsigned short* ps = Ps[wave];
    #pragma unroll
    for (int n = 0; n < 4; n++)
      #pragma unroll
      for (int r = 0; r < 8; r++)
        ps[(hl * 8 + r) * KVT + n * 16 + l15] = f2bf(sc[n][r]);
    __builtin_amdgcn_wave_barrier();
    asm volatile("s_wait_dscnt 0" ::: "memory");      // DS in-order per wave

    v16bf ap[2];
    #pragma unroll
    for (int kk = 0; kk < 2; kk++) {
      int b = l15 * KVT + kk * 32 + hl * 8;
      ap[kk] = mkfrag(*(const u16x8*)&ps[b], *(const u16x8*)&ps[b + 16]);
    }
    __builtin_amdgcn_wave_barrier();

    // ---- O += P @ V  (Vs is [e][kv]: contiguous B-fragment loads) ----
    #pragma unroll
    for (int tt = 0; tt < 4; tt++)
      #pragma unroll
      for (int kk = 0; kk < 2; kk++) {
        const unsigned short* vp = &Vs[buf][(tt * 16 + l15) * KVT + kk * 32 + hl * 8];
        v16bf bv = mkfrag(*(const u16x8*)vp, *(const u16x8*)(vp + 16));
        o[tt] = __builtin_amdgcn_wmma_f32_16x16x32_bf16(false, ap[kk], false, bv,
                                                        (short)0, o[tt], false, false);
      }
  }

  // ---- epilogue: out[s, h*64+e] = O / l ----
  #pragma unroll
  for (int r = 0; r < 8; r++) {
    float inv = 1.0f / l[r];
    #pragma unroll
    for (int tt = 0; tt < 4; tt++) {
      int col = h * HD + tt * 16 + l15;
      out[(size_t)(rowbase + r) * D_MODEL + col] = o[tt][r] * inv;
    }
  }
}

// ---------------------------------------------------------------------------
extern "C" void kernel_launch(void* const* d_in, const int* in_sizes, int n_in,
                              void* d_out, int out_size, void* d_ws, size_t ws_size,
                              hipStream_t stream) {
  const float* x    = (const float*)d_in[0];   // [1,4096,1024]
  const float* mask = (const float*)d_in[1];   // [1,1,4096,4096]
  const float* wqkv = (const float*)d_in[2];   // [1024,3072]
  float* out = (float*)d_out;                  // [1,4096,1024]

  const size_t perTensor = (size_t)NH * S_LEN * HD;          // bf16 elements
  unsigned short* Qb = (unsigned short*)d_ws;                // [H][S][E]
  unsigned short* Kb = Qb + perTensor;                       // [H][S][E]
  unsigned short* Vt = Kb + perTensor;                       // [H][E][S]
  (void)in_sizes; (void)n_in; (void)out_size; (void)ws_size;

  dim3 g1(N3 / 64, S_LEN / 128);               // 48 x 32 tiles
  qkv_gemm_kernel<<<g1, 256, 0, stream>>>(x, wqkv, Qb, Kb, Vt);

  dim3 g2(S_LEN / 64, NH);                     // 64 q-tiles x 16 heads
  flash_attn_kernel<<<g2, 128, 0, stream>>>(Qb, Kb, Vt, mask, out);
}
